// GraphBertEncoder_24773371364096
// MI455X (gfx1250) — matrix operand order repaired and checked by reading, not compiled, not run on hardware
//
#include <hip/hip_runtime.h>
#include <hip/hip_bf16.h>
#include <math.h>
#include <stdint.h>

// ---------------------------------------------------------------------------
// GraphBERT encoder for MI455X (gfx1250).
// - bf16 data at rest (weights pre-transposed+converted once; activations
//   written bf16 by producers), fp32 LayerNorm master copy in d_out.
// - All GEMM/attention tile staging via GLOBAL_LOAD_ASYNC_TO_LDS_B128
//   (uniform fixed-count loops -> no EXEC guards).
// - WMMA bf16 (f32 accum); fragment loads batched per K-step so WMMAs issue
//   back-to-back behind a single s_wait_dscnt.
// ---------------------------------------------------------------------------

typedef __attribute__((ext_vector_type(16))) __bf16 v16bf;
typedef __attribute__((ext_vector_type(8)))  __bf16 v8bf;
typedef __attribute__((ext_vector_type(4)))  __bf16 v4bf;
typedef __attribute__((ext_vector_type(8)))  float  v8f;
typedef __attribute__((ext_vector_type(4)))  float  v4f;

constexpr int   Dc   = 768;
constexpr int   Hc   = 12;
constexpr int   FFc  = 3072;
constexpr int   NLc  = 4;
constexpr int   Bc   = 8;
constexpr int   SNc  = 32;
constexpr int   Lc   = 128;
constexpr int   Nc   = Bc * SNc;          // 256 sequences
constexpr int   DHc  = Dc / Hc;           // 64
constexpr float SCALEc = 0.125f;          // 1/sqrt(64)
constexpr size_t Mtok = (size_t)Nc * Lc;  // 32768 tokens

__device__ __forceinline__ float gelu_f(float x) {
  return 0.5f * x * (1.0f + erff(x * 0.7071067811865475f));
}

__device__ __forceinline__ v8f zero_v8f() {
  v8f z;
#pragma unroll
  for (int i = 0; i < 8; ++i) z[i] = 0.0f;
  return z;
}

__device__ __forceinline__ v8bf load8bf(const __bf16* p) {       // ds_load_b128
  return *(const v8bf*)p;
}
__device__ __forceinline__ v16bf join8(v8bf a, v8bf b) {
  return __builtin_shufflevector(a, b, 0, 1, 2, 3, 4, 5, 6, 7,
                                       8, 9, 10, 11, 12, 13, 14, 15);
}
__device__ __forceinline__ v16bf load16bf(const __bf16* p) {     // 2x ds_load_b128
  return join8(load8bf(p), load8bf(p + 8));
}

// Async global->LDS 16B copy (per-lane addresses), tracked by ASYNCcnt.
__device__ __forceinline__ void async_copy_b128(void* lds, const void* g) {
  asm volatile("global_load_async_to_lds_b128 %0, %1, off"
               :: "v"((uint32_t)(uintptr_t)lds), "v"(g)
               : "memory");
}
__device__ __forceinline__ void wait_async0() {
  asm volatile("s_wait_asynccnt 0" ::: "memory");
}

// ---------------------------------------------------------------------------
// WMMA GEMM: out = act(A[M,K] @ W[K,N] + bias); A bf16 row-major, W given
// pre-transposed bf16 Wt[N][K]. 256 thr (8 waves) -> 128x64 tile, K staged 64.
// Both LDS tiles filled with async b128 copies. Output fp32 or bf16.
// M%128==0, N%64==0, K%64==0.
// ---------------------------------------------------------------------------
constexpr int GT_M = 128;
constexpr int GT_N = 64;
constexpr int GT_K = 64;
constexpr int LDA_S = GT_K + 8;   // 72 bf16 (144 B rows: 16B-aligned, no conflicts)
constexpr int LDBT  = GT_K + 8;   // 72 bf16

__global__ __launch_bounds__(256)
void gemm_bf16(const __bf16* __restrict__ A, const __bf16* __restrict__ Wt,
               const float* __restrict__ bias,
               float* __restrict__ Cf, __bf16* __restrict__ Ch,
               int Kdim, int Nout, int act, int out16)
{
  __shared__ __bf16 As[GT_M * LDA_S];   // [row][k]
  __shared__ __bf16 Bt[GT_N * LDBT];    // [col n][k]

  const int tid  = threadIdx.x;
  const int wave = tid >> 5;
  const int lane = tid & 31;
  const int l    = lane & 15;
  const int hi   = lane >> 4;
  const size_t m0 = (size_t)blockIdx.x * GT_M;
  const int    n0 = blockIdx.y * GT_N;

  v8f acc[4];
#pragma unroll
  for (int c = 0; c < 4; ++c) acc[c] = zero_v8f();

  const __bf16* arow_p = &As[(wave * 16 + l) * LDA_S];

  for (int k0 = 0; k0 < Kdim; k0 += GT_K) {
    // A tile 128x64 bf16: 4 x 16B async copies per thread (uniform count).
#pragma unroll
    for (int t = 0; t < (GT_M * GT_K) / (8 * 256); ++t) {
      int i8 = (tid + t * 256) * 8;
      int r = i8 >> 6, c = i8 & 63;
      async_copy_b128(&As[r * LDA_S + c],
                      &A[(m0 + r) * (size_t)Kdim + k0 + c]);
    }
    // W tile (Wt rows): 2 x 16B async copies per thread.
#pragma unroll
    for (int t = 0; t < (GT_N * GT_K) / (8 * 256); ++t) {
      int i8 = (tid + t * 256) * 8;
      int r = i8 >> 6, c = i8 & 63;   // r = n, c = k
      async_copy_b128(&Bt[r * LDBT + c],
                      &Wt[(size_t)(n0 + r) * Kdim + k0 + c]);
    }
    wait_async0();
    __syncthreads();

#pragma unroll
    for (int ks = 0; ks < GT_K; ks += 32) {
      v16bf af = join8(load8bf(arow_p + ks + hi * 8),
                       load8bf(arow_p + ks + 16 + hi * 8));
      v16bf wf[4];
#pragma unroll
      for (int cb = 0; cb < 4; ++cb)
        wf[cb] = join8(load8bf(&Bt[(cb * 16 + l) * LDBT + ks + hi * 8]),
                       load8bf(&Bt[(cb * 16 + l) * LDBT + ks + 16 + hi * 8]));
#pragma unroll
      for (int cb = 0; cb < 4; ++cb)
        acc[cb] = __builtin_amdgcn_wmma_f32_16x16x32_bf16(
            false, af, false, wf[cb], (short)0, acc[cb], false, false);
    }
    __syncthreads();
  }

#pragma unroll
  for (int cb = 0; cb < 4; ++cb) {
    const int col  = n0 + cb * 16 + l;
    const float bv = bias[col];
#pragma unroll
    for (int j = 0; j < 8; ++j) {
      const size_t row = m0 + wave * 16 + hi * 8 + j;
      float x = acc[cb][j] + bv;
      if (act) x = gelu_f(x);
      if (out16) Ch[row * (size_t)Nout + col] = (__bf16)x;
      else       Cf[row * (size_t)Nout + col] = x;
    }
  }
}

// ---------------------------------------------------------------------------
// Fused attention: one workgroup per (n,h). Q/K staged via async b128 copies;
// V transposed manually with batched loads. Scores + softmax in LDS; ctx bf16.
// Dynamic LDS: Qs 18432 + Ks 18432 + Vt 17408 + Ss 67584 = 121856 B.
// ---------------------------------------------------------------------------
constexpr int ATT_LDQ = DHc + 8;    // 72 bf16
constexpr int ATT_LDV = Lc + 8;     // 136 bf16
constexpr int ATT_LDS = Lc + 4;     // 132 f32
constexpr int SMEM_ATT = (2 * Lc * ATT_LDQ + DHc * ATT_LDV) * 2
                       + Lc * ATT_LDS * 4;

__global__ __launch_bounds__(256)
void attention_fused(const __bf16* __restrict__ Q, const __bf16* __restrict__ K,
                     const __bf16* __restrict__ V,
                     const float* __restrict__ rel_pos,
                     const float* __restrict__ amask,
                     __bf16* __restrict__ CTX, int first_layer)
{
  extern __shared__ char smem[];
  __bf16* Qs = (__bf16*)smem;                 // [q][d]   Lc x ATT_LDQ
  __bf16* Ks = Qs + Lc * ATT_LDQ;             // [kv][d]  Lc x ATT_LDQ
  __bf16* Vt = Ks + Lc * ATT_LDQ;             // [d][kv]  DHc x ATT_LDV
  float*  Ss = (float*)(Vt + DHc * ATT_LDV);  // [q][kv]  Lc x ATT_LDS

  const int nh = blockIdx.x;
  const int n  = nh / Hc;
  const int h  = nh % Hc;
  const int tid  = threadIdx.x;
  const int wave = tid >> 5;
  const int lane = tid & 31;
  const int l    = lane & 15;
  const int hi   = lane >> 4;

  const size_t gbase = (size_t)n * Lc * Dc + (size_t)h * DHc;   // bf16 elems

  // Q/K: pure async copies, fixed 4 chunks per thread per matrix.
#pragma unroll
  for (int t = 0; t < (Lc * DHc) / (8 * 256); ++t) {
    int i8 = (tid + t * 256) * 8;
    int r = i8 >> 6, c = i8 & 63;
    size_t g = gbase + (size_t)r * Dc + c;
    async_copy_b128(&Qs[r * ATT_LDQ + c], &Q[g]);
    async_copy_b128(&Ks[r * ATT_LDQ + c], &K[g]);
  }
  // V transpose: batch all loads first, then scatter to Vt.
  {
    v4bf vtmp[8];
    int rr[8], cc[8];
#pragma unroll
    for (int t = 0; t < 8; ++t) {
      int i4 = (tid + t * 256) * 4;
      int r = i4 >> 6, c = i4 & 63;
      rr[t] = r; cc[t] = c;
      vtmp[t] = *(const v4bf*)&V[gbase + (size_t)r * Dc + c];
    }
#pragma unroll
    for (int t = 0; t < 8; ++t)
#pragma unroll
      for (int j = 0; j < 4; ++j)
        Vt[(cc[t] + j) * ATT_LDV + rr[t]] = vtmp[t][j];
  }
  wait_async0();
  __syncthreads();

  const int arow = wave * 16 + l;
  const __bf16* qrow_p = &Qs[arow * ATT_LDQ];

  // S = Q K^T : batch 8 B-fragments per K-step, then 8 WMMAs back-to-back.
  {
    v8f sacc[8];
#pragma unroll
    for (int cb = 0; cb < 8; ++cb) sacc[cb] = zero_v8f();
#pragma unroll
    for (int k0 = 0; k0 < DHc; k0 += 32) {
      v16bf af = join8(load8bf(qrow_p + k0 + hi * 8),
                       load8bf(qrow_p + k0 + 16 + hi * 8));
      v16bf kf[8];
#pragma unroll
      for (int cb = 0; cb < 8; ++cb)
        kf[cb] = join8(load8bf(&Ks[(cb * 16 + l) * ATT_LDQ + k0 + hi * 8]),
                       load8bf(&Ks[(cb * 16 + l) * ATT_LDQ + k0 + 16 + hi * 8]));
#pragma unroll
      for (int cb = 0; cb < 8; ++cb)
        sacc[cb] = __builtin_amdgcn_wmma_f32_16x16x32_bf16(
            false, af, false, kf[cb], (short)0, sacc[cb], false, false);
    }
#pragma unroll
    for (int cb = 0; cb < 8; ++cb) {
#pragma unroll
      for (int j = 0; j < 8; ++j) {
        const int q  = wave * 16 + hi * 8 + j;
        const int kk = cb * 16 + l;
        float mv = amask[(size_t)n * Lc + kk];
        if (first_layer && ((n & (SNc - 1)) == 0) && kk == 0) mv = -10000.0f;
        const float rp = rel_pos[(((size_t)n * Hc + h) * Lc + q) * Lc + kk];
        Ss[q * ATT_LDS + kk] = sacc[cb][j] * SCALEc + rp + mv;
      }
    }
  }
  __syncthreads();

  // Row softmax.
  if (tid < Lc) {
    float* row = Ss + tid * ATT_LDS;
    float mx = -3.0e38f;
    for (int kk = 0; kk < Lc; ++kk) mx = fmaxf(mx, row[kk]);
    float sum = 0.0f;
    for (int kk = 0; kk < Lc; ++kk) { float e = __expf(row[kk] - mx); row[kk] = e; sum += e; }
    const float inv = 1.0f / sum;
    for (int kk = 0; kk < Lc; ++kk) row[kk] *= inv;
  }
  __syncthreads();

  // ctx = P @ V : batch 4 V-fragments per K-step.
  {
    v8f cacc[4];
#pragma unroll
    for (int cb = 0; cb < 4; ++cb) cacc[cb] = zero_v8f();
#pragma unroll
    for (int k0 = 0; k0 < Lc; k0 += 32) {
      const float* prow = &Ss[arow * ATT_LDS + k0 + hi * 8];
      v4f p0 = *(const v4f*)(prow);
      v4f p1 = *(const v4f*)(prow + 4);
      v4f p2 = *(const v4f*)(prow + 16);
      v4f p3 = *(const v4f*)(prow + 20);
      v16bf pf;
#pragma unroll
      for (int e = 0; e < 4; ++e) {
        pf[e]      = (__bf16)p0[e];
        pf[4 + e]  = (__bf16)p1[e];
        pf[8 + e]  = (__bf16)p2[e];
        pf[12 + e] = (__bf16)p3[e];
      }
      v16bf vf[4];
#pragma unroll
      for (int cb = 0; cb < 4; ++cb)
        vf[cb] = join8(load8bf(&Vt[(cb * 16 + l) * ATT_LDV + k0 + hi * 8]),
                       load8bf(&Vt[(cb * 16 + l) * ATT_LDV + k0 + 16 + hi * 8]));
#pragma unroll
      for (int cb = 0; cb < 4; ++cb)
        cacc[cb] = __builtin_amdgcn_wmma_f32_16x16x32_bf16(
            false, pf, false, vf[cb], (short)0, cacc[cb], false, false);
    }
#pragma unroll
    for (int cb = 0; cb < 4; ++cb) {
#pragma unroll
      for (int j = 0; j < 8; ++j) {
        const int q    = wave * 16 + hi * 8 + j;
        const int dcol = cb * 16 + l;
        CTX[((size_t)n * Lc + q) * Dc + h * DHc + dcol] = (__bf16)cacc[cb][j];
      }
    }
  }
}

// ---------------------------------------------------------------------------
// y = LayerNorm(x + res) * g + b; optional bf16 shadow copy y16.
// ---------------------------------------------------------------------------
__global__ __launch_bounds__(256)
void add_ln(const float* __restrict__ x, int ldx,
            const float* __restrict__ res, int ldres,
            const float* __restrict__ g, const float* __restrict__ b,
            float* __restrict__ y, int ldy, __bf16* __restrict__ y16)
{
  __shared__ float red[256];
  const int r = blockIdx.x;
  const float* xr = x + (size_t)r * ldx;
  const float* rr = res + (size_t)r * ldres;
  float* yr = y + (size_t)r * ldy;

  float s = 0.0f;
  for (int i = threadIdx.x; i < Dc; i += 256) s += xr[i] + rr[i];
  red[threadIdx.x] = s; __syncthreads();
  for (int o = 128; o > 0; o >>= 1) {
    if (threadIdx.x < o) red[threadIdx.x] += red[threadIdx.x + o];
    __syncthreads();
  }
  const float mu = red[0] * (1.0f / Dc);
  __syncthreads();
  float v = 0.0f;
  for (int i = threadIdx.x; i < Dc; i += 256) { float d = xr[i] + rr[i] - mu; v += d * d; }
  red[threadIdx.x] = v; __syncthreads();
  for (int o = 128; o > 0; o >>= 1) {
    if (threadIdx.x < o) red[threadIdx.x] += red[threadIdx.x + o];
    __syncthreads();
  }
  const float rstd = rsqrtf(red[0] * (1.0f / Dc) + 1e-12f);
  for (int i = threadIdx.x; i < Dc; i += 256) {
    float val = (xr[i] + rr[i] - mu) * rstd * g[i] + b[i];
    yr[i] = val;
    if (y16) y16[(size_t)r * ldy + i] = (__bf16)val;
  }
}

// ---------------------------------------------------------------------------
// One-time weight prep: Wt[n*K + k] = (bf16) W[k*N + n].
// ---------------------------------------------------------------------------
__global__ void transpose_cvt(const float* __restrict__ W, __bf16* __restrict__ Wt,
                              int K, int N)
{
  size_t id = (size_t)blockIdx.x * blockDim.x + threadIdx.x;
  if (id >= (size_t)K * N) return;
  int n = (int)(id / K), k = (int)(id % K);
  Wt[id] = (__bf16)W[(size_t)k * N + n];
}

// ---------------------------------------------------------------------------
// Small helpers (graph aggregation path, <0.1% of FLOPs).
// ---------------------------------------------------------------------------
__global__ void small_dense(const float* __restrict__ A, int lda,
                            const float* __restrict__ W,
                            const float* __restrict__ bias,
                            float* __restrict__ C,
                            int M2, int K2, int N2, int act)
{
  int id = blockIdx.x * blockDim.x + threadIdx.x;
  if (id >= M2 * N2) return;
  int m = id / N2, nn = id % N2;
  float s = bias[nn];
  const float* ar = A + (size_t)m * lda;
  for (int k = 0; k < K2; ++k) s += ar[k] * W[(size_t)k * N2 + nn];
  if (act) s = gelu_f(s);
  C[id] = s;
}

__global__ void extract_cls(const float* __restrict__ X, float* __restrict__ out,
                            __bf16* __restrict__ out16)
{
  int id = blockIdx.x * blockDim.x + threadIdx.x;
  if (id >= Nc * Dc) return;
  int row = id / Dc, d = id % Dc;                 // row = b*SN + s
  float v = X[((size_t)row * Lc + 1) * Dc + d];   // token index 1
  out[id] = v;
  out16[id] = (__bf16)v;
}

__global__ void graph_attn(const float* __restrict__ gq,
                           const __bf16* __restrict__ gk,
                           const __bf16* __restrict__ gv,
                           const float* __restrict__ nrelpos,
                           const float* __restrict__ nmask,
                           float* __restrict__ gctx)
{
  const int b = blockIdx.x / Hc;
  const int h = blockIdx.x % Hc;
  const int s = threadIdx.x;  // 0..31
  __shared__ float sc[SNc];
  const float*  qv = gq + (size_t)b * Dc + h * DHc;
  const __bf16* kv = gk + (size_t)(b * SNc + s) * Dc + h * DHc;
  float dot = 0.0f;
  for (int d = 0; d < DHc; ++d) dot += qv[d] * (float)kv[d];
  sc[s] = dot * SCALEc + nrelpos[((size_t)b * Hc + h) * SNc + s] + nmask[b * SNc + s];
  __syncthreads();
  if (s == 0) {
    float mx = -3.0e38f;
    for (int i = 0; i < SNc; ++i) mx = fmaxf(mx, sc[i]);
    float sum = 0.0f;
    for (int i = 0; i < SNc; ++i) { sc[i] = __expf(sc[i] - mx); sum += sc[i]; }
    float inv = 1.0f / sum;
    for (int i = 0; i < SNc; ++i) sc[i] *= inv;
  }
  __syncthreads();
  for (int d = s; d < DHc; d += 32) {
    float acc = 0.0f;
    for (int i = 0; i < SNc; ++i)
      acc += sc[i] * (float)gv[(size_t)(b * SNc + i) * Dc + h * DHc + d];
    gctx[(size_t)b * Dc + h * DHc + d] = acc;
  }
}

__global__ void station_scatter(const float* __restrict__ sta,
                                const float* __restrict__ act_st,
                                float* __restrict__ X, __bf16* __restrict__ X16)
{
  int id = blockIdx.x * blockDim.x + threadIdx.x;
  if (id >= Bc * Dc) return;
  int b = id / Dc, d = id % Dc;
  float v = sta[id] * act_st[b];
  X[(size_t)b * SNc * Lc * Dc + d] = v;          // hs4[b,0,0,d]
  X16[(size_t)b * SNc * Lc * Dc + d] = (__bf16)v;
}

__global__ void copy_cvt(const float* __restrict__ a, float* __restrict__ b,
                         __bf16* __restrict__ b16, size_t nelem)
{
  size_t i = (size_t)blockIdx.x * blockDim.x + threadIdx.x;
  if (i < nelem) { float v = a[i]; b[i] = v; b16[i] = (__bf16)v; }
}

// ---------------------------------------------------------------------------
// Host orchestration.
// ---------------------------------------------------------------------------
struct Blk {
  const float *ao_b, *ao_w, *inter_b, *inter_w, *k_b, *k_w;
  const float *ln1_b, *ln1_g, *ln2_b, *ln2_g;
  const float *out_b, *out_w, *q_b, *q_w, *v_b, *v_w;
};
struct BlkW { __bf16 *q, *k, *v, *ao, *inter, *out; };

extern "C" void kernel_launch(void* const* d_in, const int* in_sizes, int n_in,
                              void* d_out, int out_size, void* d_ws, size_t ws_size,
                              hipStream_t stream)
{
  (void)in_sizes; (void)n_in; (void)out_size; (void)ws_size;

  const float* hs_in   = (const float*)d_in[0];
  const float* amask   = (const float*)d_in[1];
  const float* nmask   = (const float*)d_in[2];
  const float* nrelpos = (const float*)d_in[3];
  const float* relpos  = (const float*)d_in[4];
  const float* act_st  = (const float*)d_in[5];

  // Params: pytree leaves, sorted dict keys ('graph' then 'layers').
  int idx = 6;
  auto grab = [&]() -> Blk {
    Blk bk;
    bk.ao_b    = (const float*)d_in[idx++]; bk.ao_w    = (const float*)d_in[idx++];
    bk.inter_b = (const float*)d_in[idx++]; bk.inter_w = (const float*)d_in[idx++];
    bk.k_b     = (const float*)d_in[idx++]; bk.k_w     = (const float*)d_in[idx++];
    bk.ln1_b   = (const float*)d_in[idx++]; bk.ln1_g   = (const float*)d_in[idx++];
    bk.ln2_b   = (const float*)d_in[idx++]; bk.ln2_g   = (const float*)d_in[idx++];
    bk.out_b   = (const float*)d_in[idx++]; bk.out_w   = (const float*)d_in[idx++];
    bk.q_b     = (const float*)d_in[idx++]; bk.q_w     = (const float*)d_in[idx++];
    bk.v_b     = (const float*)d_in[idx++]; bk.v_w     = (const float*)d_in[idx++];
    return bk;
  };
  Blk gp = grab();
  Blk lp[NLc];
  for (int i = 0; i < NLc; ++i) lp[i] = grab();

  // ---- workspace carve (256B aligned) ----
  char* wsp = (char*)d_ws;
  auto alloc = [&](size_t bytes) -> char* {
    char* r = wsp; wsp += (bytes + 255) & ~(size_t)255; return r;
  };
  const size_t MD = Mtok * Dc;
  __bf16* X16   = (__bf16*)alloc(MD * 2);
  __bf16* Q16   = (__bf16*)alloc(MD * 2);
  __bf16* K16   = (__bf16*)alloc(MD * 2);
  __bf16* V16   = (__bf16*)alloc(MD * 2);
  __bf16* CTX16 = (__bf16*)alloc(MD * 2);
  __bf16* Y16   = (__bf16*)alloc(MD * 2);
  float*  Yf    = (float*)alloc(MD * 4);
  float*  Tf    = (float*)alloc(MD * 4);
  __bf16* FF16  = (__bf16*)alloc(Mtok * FFc * 2);

  const size_t WDD = (size_t)Dc * Dc, WDF = (size_t)Dc * FFc;
  BlkW gw, lw[NLc];
  auto allocW = [&]() -> BlkW {
    BlkW w;
    w.q  = (__bf16*)alloc(WDD * 2); w.k   = (__bf16*)alloc(WDD * 2);
    w.v  = (__bf16*)alloc(WDD * 2); w.ao  = (__bf16*)alloc(WDD * 2);
    w.inter = (__bf16*)alloc(WDF * 2); w.out = (__bf16*)alloc(WDF * 2);
    return w;
  };
  gw = allocW();
  for (int i = 0; i < NLc; ++i) lw[i] = allocW();

  float*  g_cls   = (float*)alloc((size_t)Nc * Dc * 4);
  __bf16* g_cls16 = (__bf16*)alloc((size_t)Nc * Dc * 2);
  __bf16* g_k16   = (__bf16*)alloc((size_t)Nc * Dc * 2);
  __bf16* g_v16   = (__bf16*)alloc((size_t)Nc * Dc * 2);
  float*  g_q   = (float*)alloc((size_t)Bc * Dc * 4);
  float*  g_ctx = (float*)alloc((size_t)Bc * Dc * 4);
  float*  g_t   = (float*)alloc((size_t)Bc * Dc * 4);
  float*  g_att = (float*)alloc((size_t)Bc * Dc * 4);
  float*  g_int = (float*)alloc((size_t)Bc * FFc * 4);
  float*  g_t2  = (float*)alloc((size_t)Bc * Dc * 4);
  float*  g_sta = (float*)alloc((size_t)Bc * Dc * 4);

  // ---- one-time weight transpose+convert (runs each call; tiny) ----
  auto conv = [&](const float* w, __bf16* wt, int K, int N) {
    transpose_cvt<<<(unsigned)(((size_t)K * N + 255) / 256), 256, 0, stream>>>(w, wt, K, N);
  };
  auto convBlk = [&](const Blk& p, const BlkW& w) {
    conv(p.q_w, w.q, Dc, Dc);       conv(p.k_w, w.k, Dc, Dc);
    conv(p.v_w, w.v, Dc, Dc);       conv(p.ao_w, w.ao, Dc, Dc);
    conv(p.inter_w, w.inter, Dc, FFc);
    conv(p.out_w, w.out, FFc, Dc);
  };
  convBlk(gp, gw);
  for (int i = 0; i < NLc; ++i) convBlk(lp[i], lw[i]);

  float* X = (float*)d_out;
  copy_cvt<<<(unsigned)((MD + 255) / 256), 256, 0, stream>>>(hs_in, X, X16, MD);

  for (int li = 0; li < NLc; ++li) {
    if (li > 0) {
      // ---- graph aggregation (tiny) ----
      extract_cls<<<(Nc * Dc + 255) / 256, 256, 0, stream>>>(X, g_cls, g_cls16);
      small_dense<<<(Bc * Dc + 255) / 256, 256, 0, stream>>>(
          g_cls, SNc * Dc, gp.q_w, gp.q_b, g_q, Bc, Dc, Dc, 0);
      {
        dim3 gg(Nc / GT_M, Dc / GT_N);
        gemm_bf16<<<gg, 256, 0, stream>>>(g_cls16, gw.k, gp.k_b,
                                          nullptr, g_k16, Dc, Dc, 0, 1);
        gemm_bf16<<<gg, 256, 0, stream>>>(g_cls16, gw.v, gp.v_b,
                                          nullptr, g_v16, Dc, Dc, 0, 1);
      }
      graph_attn<<<Bc * Hc, 32, 0, stream>>>(g_q, g_k16, g_v16, nrelpos, nmask, g_ctx);
      small_dense<<<(Bc * Dc + 255) / 256, 256, 0, stream>>>(
          g_ctx, Dc, gp.ao_w, gp.ao_b, g_t, Bc, Dc, Dc, 0);
      add_ln<<<Bc, 256, 0, stream>>>(g_t, Dc, g_cls, SNc * Dc,
                                     gp.ln1_g, gp.ln1_b, g_att, Dc, nullptr);
      small_dense<<<(Bc * FFc + 255) / 256, 256, 0, stream>>>(
          g_att, Dc, gp.inter_w, gp.inter_b, g_int, Bc, Dc, FFc, 1);
      small_dense<<<(Bc * Dc + 255) / 256, 256, 0, stream>>>(
          g_int, FFc, gp.out_w, gp.out_b, g_t2, Bc, FFc, Dc, 0);
      add_ln<<<Bc, 256, 0, stream>>>(g_t2, Dc, g_att, Dc,
                                     gp.ln2_g, gp.ln2_b, g_sta, Dc, nullptr);
      station_scatter<<<(Bc * Dc + 255) / 256, 256, 0, stream>>>(g_sta, act_st, X, X16);
    }

    // ---- BERT layer (all bf16 data movement) ----
    const Blk&  P  = lp[li];
    const BlkW& Wb = lw[li];
    dim3 gq((unsigned)(Mtok / GT_M), Dc / GT_N);    // 256 x 12
    gemm_bf16<<<gq, 256, 0, stream>>>(X16, Wb.q, P.q_b, nullptr, Q16, Dc, Dc, 0, 1);
    gemm_bf16<<<gq, 256, 0, stream>>>(X16, Wb.k, P.k_b, nullptr, K16, Dc, Dc, 0, 1);
    gemm_bf16<<<gq, 256, 0, stream>>>(X16, Wb.v, P.v_b, nullptr, V16, Dc, Dc, 0, 1);

    attention_fused<<<Nc * Hc, 256, SMEM_ATT, stream>>>(
        Q16, K16, V16, relpos, amask, CTX16, li == 0 ? 1 : 0);

    gemm_bf16<<<gq, 256, 0, stream>>>(CTX16, Wb.ao, P.ao_b, Tf, nullptr, Dc, Dc, 0, 0);
    add_ln<<<(unsigned)Mtok, 256, 0, stream>>>(Tf, Dc, X, Dc,
                                               P.ln1_g, P.ln1_b, Yf, Dc, Y16);

    dim3 gi((unsigned)(Mtok / GT_M), FFc / GT_N);   // 256 x 48
    gemm_bf16<<<gi, 256, 0, stream>>>(Y16, Wb.inter, P.inter_b,
                                      nullptr, FF16, Dc, FFc, 1, 1);
    gemm_bf16<<<gq, 256, 0, stream>>>(FF16, Wb.out, P.out_b, Tf, nullptr, FFc, Dc, 0, 0);
    add_ln<<<(unsigned)Mtok, 256, 0, stream>>>(Tf, Dc, Yf, Dc,
                                               P.ln2_g, P.ln2_b, X, Dc, X16);
  }
}
